// ABCDSpatialModule_88776974009046
// MI455X (gfx1250) — matrix operand order repaired and checked by reading, not compiled, not run on hardware
//
#include <hip/hip_runtime.h>
#include <hip/hip_bf16.h>
#include <math.h>

#define BB 8
#define TT 12
#define NN 2048
#define LL 96
#define HD 64
#define HA 32
#define KT 64
#define NTILES (NN/16)
#define TOPK 20

typedef __attribute__((ext_vector_type(16))) _Float16 v16h;
typedef __attribute__((ext_vector_type(8)))  float    v8f;
typedef __attribute__((ext_vector_type(4)))  unsigned v4u;
typedef __attribute__((ext_vector_type(8)))  int      v8i;
typedef __attribute__((ext_vector_type(4)))  int      v4i;

// monotone float -> uint mapping (total order matches float compare)
__device__ __forceinline__ unsigned f2ord(float f){
  unsigned u = __float_as_uint(f);
  return (u & 0x80000000u) ? ~u : (u | 0x80000000u);
}
__device__ __forceinline__ float wred_sum(float v){
#pragma unroll
  for (int m=16;m;m>>=1) v += __shfl_xor(v,m,32);
  return v;
}
__device__ __forceinline__ int wred_sumi(int v){
#pragma unroll
  for (int m=16;m;m>>=1) v += __shfl_xor(v,m,32);
  return v;
}
__device__ __forceinline__ float wred_max(float v){
#pragma unroll
  for (int m=16;m;m>>=1) v = fmaxf(v,__shfl_xor(v,m,32));
  return v;
}

// ---------------- kernel 1: static row sums + hybrid softmax ----------------
__global__ void k_prep(const float* __restrict__ adj, const float* __restrict__ hlog,
                       float* __restrict__ invRow, float* __restrict__ wmix){
  if (blockIdx.x == NN){
    if (threadIdx.x==0){
      float a=hlog[0], b=hlog[1], c=hlog[2];
      float m=fmaxf(a,fmaxf(b,c));
      float ea=__expf(a-m), eb=__expf(b-m), ec=__expf(c-m);
      float s=ea+eb+ec;
      wmix[0]=ea/s; wmix[1]=eb/s; wmix[2]=ec/s;
    }
    return;
  }
  __shared__ float red[256];
  int row = blockIdx.x;
  float s=0.f;
  for (int j=threadIdx.x;j<NN;j+=256) s += adj[(size_t)row*NN + j];
  red[threadIdx.x]=s; __syncthreads();
  for (int st=128; st; st>>=1){ if (threadIdx.x<st) red[threadIdx.x]+=red[threadIdx.x+st]; __syncthreads(); }
  if (threadIdx.x==0) invRow[row] = 1.0f/fmaxf(red[0],1e-6f);
}

// ---------------- kernel 2: q/k projection + l2norm + WMMA fragment packing -
// qfrag: A-operand layout (16x32 f16): per 16-row tile, 2 K-fragments,
//        32 lanes x 16 halves each.   kfrag: B-operand layout (32x16 f16).
__global__ void k_qk(const float* __restrict__ hist, const float* __restrict__ Wq,
                     const float* __restrict__ Wk,
                     _Float16* __restrict__ qfrag, _Float16* __restrict__ kfrag){
  __shared__ float sig[LL];
  __shared__ float vals[128];
  int bn = blockIdx.x;
  int b = bn / NN, n = bn % NN;
  int t = threadIdx.x;
  if (t < LL) sig[t] = hist[((size_t)b*LL + t)*NN + n];
  __syncthreads();
  bool isQ = (t < 64);
  int h = isQ ? t : (t-64);
  const float* W = isQ ? Wq : Wk;
  float acc = 0.f;
#pragma unroll 8
  for (int l=0; l<LL; ++l) acc += sig[l]*W[h*LL + l];
  vals[t] = acc; __syncthreads();
  float ss = 0.f;
  int base = isQ ? 0 : 64;
#pragma unroll 8
  for (int i=0;i<64;++i){ float x=vals[base+i]; ss += x*x; }
  float v = acc / fmaxf(sqrtf(ss), 1e-12f);

  int tile = n >> 4, m = n & 15, frag = h >> 5, kk = h & 31;
  size_t fbase = ((size_t)((b*NTILES + tile)*2 + frag))*512;
  if (isQ){
    int grp = kk >> 3;                            // 0..3
    int lane = m + ((grp & 1) ? 16 : 0);
    int elem = (kk & 7) + ((grp >= 2) ? 8 : 0);
    qfrag[fbase + lane*16 + elem] = (_Float16)v;
  } else {
    int lane = m + (((kk >> 3) & 1) ? 16 : 0);
    int elem = (kk & 7) + ((kk >= 16) ? 8 : 0);
    kfrag[fbase + lane*16 + elem] = (_Float16)v;
  }
}

// ---------------- kernel 3: adaptive basis src/dst -------------------------
__global__ void k_srcdst(const float* __restrict__ Uk, const float* __restrict__ Zs,
                         const float* __restrict__ Zd,
                         float* __restrict__ srcb, float* __restrict__ dstb){
  __shared__ float U[KT];
  __shared__ float vv[64];
  int n = blockIdx.x, t = threadIdx.x;
  U[t] = Uk[(size_t)n*KT + t];
  __syncthreads();
  bool isS = (t < 32);
  int h = t & 31;
  const float* Z = isS ? Zs : Zd;
  float acc = 0.f;
#pragma unroll 8
  for (int k=0;k<KT;++k) acc += U[k]*Z[k*HA + h];
  vv[t] = acc; __syncthreads();
  float ss = 0.f;
  int base = isS ? 0 : 32;
#pragma unroll 8
  for (int i=0;i<32;++i){ float x=vv[base+i]; ss += x*x; }
  float o = acc / fmaxf(sqrtf(ss), 1e-12f);
  (isS ? srcb : dstb)[(size_t)n*HA + h] = o;
}

// ---------------- kernel 4: adaptive adjacency, top-20 sparse --------------
__global__ void k_adp(const float* __restrict__ srcb, const float* __restrict__ dstb,
                      int* __restrict__ aidx, float* __restrict__ aval){
  extern __shared__ float buf[];                 // [8][2048]
  __shared__ int cnt[8];
  int wv = threadIdx.x >> 5, lane = threadIdx.x & 31;
  int row = blockIdx.x*8 + wv;
  float* L = buf + wv*NN;

  float sr[HA];
#pragma unroll
  for (int h=0;h<HA;++h) sr[h] = srcb[(size_t)row*HA + h];

  for (int j=lane; j<NN; j+=32){
    float a = 0.f;
#pragma unroll
    for (int h=0;h<HA;++h) a += sr[h]*dstb[(size_t)j*HA + h];
    L[j] = a * 0.17677669529663689f;             // 1/sqrt(32)
  }

  // 20th largest via bitwise threshold bisection on ordered uints
  unsigned thr = 0u;
  for (int bit=31; bit>=0; --bit){
    unsigned cand = thr | (1u<<bit);
    int c = 0;
    for (int s=0;s<NN/32;++s) c += (f2ord(L[lane + 32*s]) >= cand);
    c = wred_sumi(c);
    if (c >= TOPK) thr = cand;
  }
  float mx = -3.4e38f;
  for (int s=0;s<NN/32;++s) mx = fmaxf(mx, L[lane + 32*s]);
  mx = wred_max(mx);
  float sum = 0.f;
  for (int s=0;s<NN/32;++s){
    float v = L[lane + 32*s];
    if (f2ord(v) >= thr) sum += __expf(2.0f*(v-mx));   // /tau, tau=0.5
  }
  sum = wred_sum(sum);

  if (lane==0) cnt[wv] = 0;
  aval[(size_t)row*32 + lane] = 0.f;
  aidx[(size_t)row*32 + lane] = 0;
  for (int s=0;s<NN/32;++s){
    int j = lane + 32*s;
    float v = L[j];
    if (f2ord(v) >= thr){
      int slot = atomicAdd(&cnt[wv], 1);
      if (slot < 32){
        aidx[(size_t)row*32 + slot] = j;
        aval[(size_t)row*32 + slot] = __expf(2.0f*(v-mx))/sum;
      }
    }
  }
}

// ---------------- kernel 5: fused dyn-logits (WMMA) + TDM adj strip +
//                  topk/softmax + blend + propagation + residual -----------
__global__ void k_main(const float* __restrict__ xin, const float* __restrict__ adj,
                       const float* __restrict__ invRow, const float* __restrict__ wmix,
                       const int* __restrict__ aidx, const float* __restrict__ aval,
                       const _Float16* __restrict__ qfrag, const _Float16* __restrict__ kfrag,
                       float* __restrict__ outp){
  extern __shared__ float shbuf[];               // [16][2048] logits + [16][2048] adj
  float* Lt_raw = shbuf;
  float* adjL   = shbuf + 16*NN;
  int wv = threadIdx.x >> 5, lane = threadIdx.x & 31;
  int blk = blockIdx.x;
  int b = blk >> 7, ti = blk & (NTILES-1);
  int i0 = ti*16;
  float w0 = wmix[0], w1 = wmix[1], w2 = wmix[2];

  // ---- TDM: async DMA of adj[i0:i0+16, 0:2048] (f32, row stride N) into LDS.
  // D# per CDNA5 ISA ch.8: group0 = {count=1 | lds_addr | global_addr(57b) | type=2}
  // group1 = {data_size=4B, tensor_dim0=2048, tensor_dim1=2048, tile_dim0=2048,
  //           tile_dim1=16, dim0_stride=2048}.  Issued once (wave 0); EXEC ignored.
  if (wv == 0){
    unsigned long long ga = (unsigned long long)(uintptr_t)(adj + (size_t)i0*NN);
    unsigned ldsOff = __builtin_amdgcn_groupstaticsize() + (unsigned)(16*NN*sizeof(float));
    v4u g0 = { 1u,                                   // count=1 (valid descriptor)
               ldsOff,                               // lds_addr (bytes)
               (unsigned)(ga & 0xFFFFFFFFull),       // global_addr[31:0]
               (unsigned)((ga >> 32) & 0x1FFFFFFull) | (2u << 30) }; // addr[56:32]|type=2
    v8i g1 = { 0x20000,                              // wg_mask=0, data_size=2 (4B)
               0x08000000,                           // tensor_dim0 = 2048 (bits 63:48)
               0x08000000,                           // tensor_dim1 = 2048 (bits 95:80)
               0x08000000,                           // tile_dim0   = 2048 (bits 127:112)
               16,                                   // tile_dim1   = 16   (bits 143:128)
               2048,                                 // tensor_dim0_stride low32
               0, 0 };
    v4i g2 = { 0, 0, 0, 0 };
    v4i g3 = { 0, 0, 0, 0 };
#if defined(__clang_major__) && (__clang_major__ >= 23)
    v8i g4 = { 0, 0, 0, 0, 0, 0, 0, 0 };
    __builtin_amdgcn_tensor_load_to_lds(g0, g1, g2, g3, g4, 0);
#else
    __builtin_amdgcn_tensor_load_to_lds(g0, g1, g2, g3, 0);
#endif
  }

  // ---- stage 1 (overlaps TDM): dyn logits strip [16 x 2048] via WMMA f16
  const _Float16* qf = qfrag + ((size_t)((b*NTILES + ti)*2))*512;
  v16h a0 = *(const v16h*)(qf + lane*16);
  v16h a1 = *(const v16h*)(qf + 512 + lane*16);
  for (int tj = wv; tj < NTILES; tj += 8){
    const _Float16* kf = kfrag + ((size_t)((b*NTILES + tj)*2))*512;
    v16h b0 = *(const v16h*)(kf + lane*16);
    v16h b1 = *(const v16h*)(kf + 512 + lane*16);
    v8f c = {};
    c = __builtin_amdgcn_wmma_f32_16x16x32_f16(false, a0, false, b0, (short)0, c, false, false);
    c = __builtin_amdgcn_wmma_f32_16x16x32_f16(false, a1, false, b1, (short)0, c, false, false);
    int col = tj*16 + (lane & 15);
    int rBase = (lane < 16) ? 0 : 8;
#pragma unroll
    for (int r=0;r<8;++r){
      Lt_raw[(rBase + r)*NN + col] = c[r]*0.125f;     // qk / sqrt(64)
    }
  }
  if (wv == 0) __builtin_amdgcn_s_wait_tensorcnt(0);
  __syncthreads();

  // ---- stage 2: inject static, per-row top-20 + softmax + hybrid blend
  for (int rr=0; rr<2; ++rr){
    int r = wv*2 + rr;
    int gi = i0 + r;
    float* L = Lt_raw + r*NN;
    const float* A = adjL + r*NN;
    float invR = invRow[gi];
    for (int s=0;s<NN/32;++s){
      int j = lane + 32*s;
      L[j] += 0.2f * A[j] * invR;                  // STATIC_W * static_adj
    }
    unsigned thr = 0u;
    for (int bit=31; bit>=0; --bit){
      unsigned cand = thr | (1u<<bit);
      int c = 0;
      for (int s=0;s<NN/32;++s) c += (f2ord(L[lane + 32*s]) >= cand);
      c = wred_sumi(c);
      if (c >= TOPK) thr = cand;
    }
    float mx = -3.4e38f;
    for (int s=0;s<NN/32;++s) mx = fmaxf(mx, L[lane + 32*s]);
    mx = wred_max(mx);
    float sum = 0.f;
    for (int s=0;s<NN/32;++s){
      float v = L[lane + 32*s];
      if (f2ord(v) >= thr) sum += __expf(2.0f*(v-mx));
    }
    sum = wred_sum(sum);
    for (int s=0;s<NN/32;++s){
      int j = lane + 32*s;
      float v = L[j];
      float dynp = (f2ord(v) >= thr) ? __expf(2.0f*(v-mx))/sum : 0.f;
      L[j] = w0*(A[j]*invR) + w2*dynp;
    }
    // scatter sparse adaptive contribution (unique idx per slot per row)
    {
      int idx = aidx[(size_t)gi*32 + lane];
      float val = aval[(size_t)gi*32 + lane];
      if (val > 0.f) L[idx] += w1*val;
    }
  }
  __syncthreads();

  // ---- stage 3: refine[b,t,i] = sum_j hyb[i,j]*x[b,t,j]; out = x + 0.2*refine
  for (int p = wv; p < TT*16; p += 8){
    int t = p >> 4, il = p & 15;
    const float* xr = xin + ((size_t)(b*TT + t))*NN;
    __builtin_prefetch(&xr[lane*64], 0, 0);
    float acc = 0.f;
    const float* L = Lt_raw + il*NN;
    for (int s=0;s<NN/32;++s){
      int j = lane + 32*s;
      acc += L[j]*xr[j];
    }
    acc = wred_sum(acc);
    if (lane == 0){
      size_t oi = ((size_t)(b*TT + t))*NN + i0 + il;
      outp[oi] = xin[oi] + 0.2f*acc;
    }
  }
}

extern "C" void kernel_launch(void* const* d_in, const int* in_sizes, int n_in,
                              void* d_out, int out_size, void* d_ws, size_t ws_size,
                              hipStream_t stream) {
  const float* x_out  = (const float*)d_in[0];   // output [B,T,N,1]
  const float* hist   = (const float*)d_in[1];   // [B,L,N]
  const float* Wq     = (const float*)d_in[2];   // [64,96]
  const float* Wk     = (const float*)d_in[3];
  const float* Zs     = (const float*)d_in[4];   // [64,32]
  const float* Zd     = (const float*)d_in[5];
  const float* Uk     = (const float*)d_in[6];   // [N,64]
  const float* hl     = (const float*)d_in[7];   // [3]
  const float* adj    = (const float*)d_in[8];   // [N,N]
  float* outp = (float*)d_out;

  char* ws = (char*)d_ws;
  size_t off = 0;
  float* wmix   = (float*)(ws+off); off += 256;
  float* invRow = (float*)(ws+off); off += (size_t)NN*4;
  float* srcb   = (float*)(ws+off); off += (size_t)NN*HA*4;
  float* dstb   = (float*)(ws+off); off += (size_t)NN*HA*4;
  int*   aidx   = (int*)  (ws+off); off += (size_t)NN*32*4;
  float* aval   = (float*)(ws+off); off += (size_t)NN*32*4;
  _Float16* qf  = (_Float16*)(ws+off); off += (size_t)BB*NTILES*2*512*2;
  _Float16* kf  = (_Float16*)(ws+off); off += (size_t)BB*NTILES*2*512*2;

  k_prep  <<<NN+1, 256, 0, stream>>>(adj, hl, invRow, wmix);
  k_qk    <<<BB*NN, 128, 0, stream>>>(hist, Wq, Wk, qf, kf);
  k_srcdst<<<NN, 64, 0, stream>>>(Uk, Zs, Zd, srcb, dstb);
  k_adp   <<<NN/8, 256, 8*NN*sizeof(float), stream>>>(srcb, dstb, aidx, aval);
  k_main  <<<BB*NTILES, 256, 2*16*NN*sizeof(float), stream>>>(
      x_out, adj, invRow, wmix, aidx, aval, qf, kf, outp);
}
